// MultiHeadAttention_38208029065977
// MI455X (gfx1250) — matrix operand order repaired and checked by reading, not compiled, not run on hardware
//
#include <hip/hip_runtime.h>

// MHA with RoPE for MI455X (gfx1250), wave32, WMMA f16->f32.
// B=2, S=2048, D=1024, H=16, Dh=64.

#define BATCH 2
#define SEQ 2048
#define DMODEL 1024
#define NHEADS 16
#define DHEAD 64
#define BS (BATCH * SEQ)          // 4096 tokens

typedef __attribute__((ext_vector_type(16))) _Float16 v16h;
typedef __attribute__((ext_vector_type(8)))  _Float16 v8h;
typedef __attribute__((ext_vector_type(4)))  _Float16 v4h;
typedef __attribute__((ext_vector_type(8)))  float    v8f;

static __device__ __forceinline__ v16h cat8(v8h lo, v8h hi) {
    return __builtin_shufflevector(lo, hi, 0,1,2,3,4,5,6,7,8,9,10,11,12,13,14,15);
}

// A-fragment (16x32 f16): row = lane%16, contiguous K per half-wave:
//   K = 16*half + e
// p points at element [row][k0] of a row-major matrix.
static __device__ __forceinline__ v16h load_afrag(const _Float16* p, int half) {
    v8h lo = *reinterpret_cast<const v8h*>(p + half * 16);
    v8h hi = *reinterpret_cast<const v8h*>(p + half * 16 + 8);
    return cat8(lo, hi);
}

static __device__ __forceinline__ v8f wmma_f16(v16h a, v16h b, v8f c) {
    return __builtin_amdgcn_wmma_f32_16x16x32_f16(false, a, false, b, (short)0, c,
                                                  false, false);
}

// ---------------------------------------------------------------------------
// f32 -> f16 conversion, 4 elements per thread
// ---------------------------------------------------------------------------
__global__ void cvt_f32_f16_k(const float* __restrict__ src,
                              _Float16* __restrict__ dst, int n4) {
    int i = blockIdx.x * 256 + threadIdx.x;
    if (i >= n4) return;
    const float4 v = reinterpret_cast<const float4*>(src)[i];
    v4h o;
    o[0] = (_Float16)v.x; o[1] = (_Float16)v.y;
    o[2] = (_Float16)v.z; o[3] = (_Float16)v.w;
    reinterpret_cast<v4h*>(dst)[i] = o;
}

// ---------------------------------------------------------------------------
// GEMM: Y[m][n] = sum_k A[m][k] * W[n][k]   (m in [0,4096), n,k in [0,1024))
// Per wave: 32 (M) x 64 (N) tile -> 8 WMMAs per 32-K step on 2 A + 4 B frags,
// with explicit double buffering (next step's frags loaded before this step's
// WMMAs are issued). Block = 8 waves = 256x64 tile. Grid (16,16).
// mode 0: f16 token-major out   mode 1: f16 V-transposed [b][h][d][S]
// mode 2: f32 token-major out
// ---------------------------------------------------------------------------
__global__ __launch_bounds__(256) void gemm_xwT_k(const _Float16* __restrict__ A,
                                                  const _Float16* __restrict__ W,
                                                  _Float16* __restrict__ outh,
                                                  float* __restrict__ outf,
                                                  int mode) {
    const int Kd = DMODEL, Nd = DMODEL;
    int w = threadIdx.x >> 5;
    int lane = threadIdx.x & 31;
    int lr = lane & 15;
    int half = lane >> 4;
    int m0 = blockIdx.x * 256 + w * 32;
    int n0 = blockIdx.y * 64;

    const _Float16* Arow0 = A + (size_t)(m0 + lr) * Kd;
    const _Float16* Arow1 = A + (size_t)(m0 + 16 + lr) * Kd;
    const _Float16* Wr[4];
#pragma unroll
    for (int j = 0; j < 4; ++j)
        Wr[j] = W + (size_t)(n0 + j * 16 + lr) * Kd + half * 16;

    v8f acc0[4] = {v8f{}, v8f{}, v8f{}, v8f{}};
    v8f acc1[4] = {v8f{}, v8f{}, v8f{}, v8f{}};

    // prologue: load K-step 0
    v16h a0c = load_afrag(Arow0, half);
    v16h a1c = load_afrag(Arow1, half);
    v16h bc[4];
#pragma unroll
    for (int j = 0; j < 4; ++j)
        bc[j] = *reinterpret_cast<const v16h*>(Wr[j]);

    for (int kk = 0; kk < Kd - 32; kk += 32) {
        int kn = kk + 32;
        __builtin_prefetch((const void*)(Arow0 + kk + 256), 0, 3);
        __builtin_prefetch((const void*)(Arow1 + kk + 256), 0, 3);
        // issue next step's loads first ...
        v16h a0n = load_afrag(Arow0 + kn, half);
        v16h a1n = load_afrag(Arow1 + kn, half);
        v16h bn[4];
#pragma unroll
        for (int j = 0; j < 4; ++j)
            bn[j] = *reinterpret_cast<const v16h*>(Wr[j] + kn);
        // ... then 8 back-to-back WMMAs on the current buffers
#pragma unroll
        for (int j = 0; j < 4; ++j) acc0[j] = wmma_f16(a0c, bc[j], acc0[j]);
#pragma unroll
        for (int j = 0; j < 4; ++j) acc1[j] = wmma_f16(a1c, bc[j], acc1[j]);
        a0c = a0n; a1c = a1n;
#pragma unroll
        for (int j = 0; j < 4; ++j) bc[j] = bn[j];
    }
    // epilogue K-step
#pragma unroll
    for (int j = 0; j < 4; ++j) acc0[j] = wmma_f16(a0c, bc[j], acc0[j]);
#pragma unroll
    for (int j = 0; j < 4; ++j) acc1[j] = wmma_f16(a1c, bc[j], acc1[j]);

    // store 32x64 tile
#pragma unroll
    for (int sub = 0; sub < 2; ++sub) {
        v8f* acc = sub ? acc1 : acc0;
#pragma unroll
        for (int j = 0; j < 4; ++j) {
            int n = n0 + j * 16 + lr;
#pragma unroll
            for (int r = 0; r < 8; ++r) {
                int m = m0 + sub * 16 + r + 8 * half;
                float v = acc[j][r];
                if (mode == 0) {
                    outh[(size_t)m * Nd + n] = (_Float16)v;
                } else if (mode == 1) {
                    int bb = m >> 11, s = m & (SEQ - 1);
                    int hh = n >> 6, d = n & 63;
                    outh[((size_t)((bb * NHEADS + hh) * DHEAD + d)) * SEQ + s] =
                        (_Float16)v;
                } else {
                    outf[(size_t)m * Nd + n] = v;
                }
            }
        }
    }
}

// ---------------------------------------------------------------------------
// RoPE on Q and K (token-major f16). Pairs are consecutive elements (2p).
// ---------------------------------------------------------------------------
__global__ void rope_k(_Float16* __restrict__ Q, _Float16* __restrict__ K,
                       const int* __restrict__ pos, int npairs) {
    int p = blockIdx.x * 256 + threadIdx.x;
    if (p >= npairs) return;
    int m = p >> 9;            // token index (D/2 = 512 pairs per token)
    int j = p & 511;
    int i = j & 31;            // pair index within head (Dh/2 = 32)
    // inv_freq = 10000^(-2i/64) = exp(-i * ln(10000)/32)
    float inv_freq = __expf(-0.28782313662f * (float)i);
    float ang = (float)pos[m] * inv_freq;
    float s, c;
    __sincosf(ang, &s, &c);
    size_t e = (size_t)p * 2;
    float q0 = (float)Q[e], q1 = (float)Q[e + 1];
    Q[e]     = (_Float16)(q0 * c - q1 * s);
    Q[e + 1] = (_Float16)(q0 * s + q1 * c);
    float k0 = (float)K[e], k1 = (float)K[e + 1];
    K[e]     = (_Float16)(k0 * c - k1 * s);
    K[e + 1] = (_Float16)(k0 * s + k1 * c);
}

// ---------------------------------------------------------------------------
// Flash attention: one wave per (b, h, 16-query tile). 32-key tiles.
// Q,K token-major f16; Vt is [b][h][d][S] f16; output Oh token-major f16.
// ---------------------------------------------------------------------------
__global__ __launch_bounds__(256) void attn_fwd_k(const _Float16* __restrict__ Q,
                                                  const _Float16* __restrict__ K,
                                                  const _Float16* __restrict__ Vt,
                                                  _Float16* __restrict__ Oh) {
    __shared__ _Float16 lds[8 * 16 * 32];   // per-wave 16x32 P tile

    int w = threadIdx.x >> 5;
    int lane = threadIdx.x & 31;
    int lr = lane & 15;
    int half = lane >> 4;

    int gw = blockIdx.x * 8 + w;            // 0 .. 4095
    int qt = gw & 127;
    int h  = (gw >> 7) & (NHEADS - 1);
    int b  = gw >> 11;
    int q0 = qt * 16;

    // Q A-frags for this query tile (K-dim = d: two frags cover d=0..63)
    const _Float16* Qrow = Q + ((size_t)(b * SEQ + q0 + lr) * DMODEL + h * DHEAD);
    v16h aq0 = load_afrag(Qrow, half);
    v16h aq1 = load_afrag(Qrow + 32, half);

    float m_i[8], l_i[8];
    v8f acc[4] = {v8f{}, v8f{}, v8f{}, v8f{}};
#pragma unroll
    for (int r = 0; r < 8; ++r) { m_i[r] = -3.0e38f; l_i[r] = 0.0f; }

    _Float16* lp = lds + w * (16 * 32);
    const _Float16* Kbase = K + ((size_t)(b * SEQ) * DMODEL + h * DHEAD);
    const _Float16* Vbase = Vt + ((size_t)((b * NHEADS + h) * DHEAD)) * SEQ;
    const float scale = 0.125f;             // 1/sqrt(64)
    int nkt = (q0 + 15) / 32 + 1;

    for (int kt = 0; kt < nkt; ++kt) {
        int kb = kt * 32;
        // prefetch next key tile's K rows / V columns
        __builtin_prefetch((const void*)(Kbase + (size_t)(kb + 32 + lr) * DMODEL), 0, 3);
        __builtin_prefetch((const void*)(Vbase + (size_t)lr * SEQ + kb + 32), 0, 3);

        // ---- scores: S = Q * K^T, two 16-key subtiles, loads grouped ----
        int keyA = kb + lr;
        int keyB = kb + 16 + lr;
        const _Float16* KrA = Kbase + (size_t)keyA * DMODEL;
        const _Float16* KrB = Kbase + (size_t)keyB * DMODEL;
        v16h bkA0 = *reinterpret_cast<const v16h*>(KrA + half * 16);
        v16h bkA1 = *reinterpret_cast<const v16h*>(KrA + 32 + half * 16);
        v16h bkB0 = *reinterpret_cast<const v16h*>(KrB + half * 16);
        v16h bkB1 = *reinterpret_cast<const v16h*>(KrB + 32 + half * 16);
        v8f s0 = v8f{}, s1 = v8f{};
        s0 = wmma_f16(aq0, bkA0, s0);
        s0 = wmma_f16(aq1, bkA1, s0);
        s1 = wmma_f16(aq0, bkB0, s1);
        s1 = wmma_f16(aq1, bkB1, s1);

#pragma unroll
        for (int r = 0; r < 8; ++r) {
            int qg = q0 + r + 8 * half;
            s0[r] = (keyA > qg) ? -3.0e38f : s0[r] * scale;
            s1[r] = (keyB > qg) ? -3.0e38f : s1[r] * scale;
        }
        // ---- online softmax (row reductions across 16-lane half-groups) --
#pragma unroll
        for (int r = 0; r < 8; ++r) {
            float v = fmaxf(s0[r], s1[r]);
            v = fmaxf(v, __shfl_xor(v, 1, 32));
            v = fmaxf(v, __shfl_xor(v, 2, 32));
            v = fmaxf(v, __shfl_xor(v, 4, 32));
            v = fmaxf(v, __shfl_xor(v, 8, 32));
            float nm = fmaxf(m_i[r], v);
            float alpha = __expf(m_i[r] - nm);
            m_i[r] = nm;
            float e0 = __expf(s0[r] - nm);
            float e1 = __expf(s1[r] - nm);
            s0[r] = e0; s1[r] = e1;
            float rs = e0 + e1;
            rs += __shfl_xor(rs, 1, 32);
            rs += __shfl_xor(rs, 2, 32);
            rs += __shfl_xor(rs, 4, 32);
            rs += __shfl_xor(rs, 8, 32);
            l_i[r] = l_i[r] * alpha + rs;
            acc[0][r] *= alpha;
            acc[1][r] *= alpha;
            acc[2][r] *= alpha;
            acc[3][r] *= alpha;
        }
        // ---- P through LDS to reshape C-layout -> A-fragment ------------
#pragma unroll
        for (int r = 0; r < 8; ++r) {
            int ql = r + 8 * half;
            lp[ql * 32 + lr]      = (_Float16)s0[r];
            lp[ql * 32 + 16 + lr] = (_Float16)s1[r];
        }
        v8h plo = *reinterpret_cast<const v8h*>(lp + lr * 32 + half * 16);
        v8h phi = *reinterpret_cast<const v8h*>(lp + lr * 32 + half * 16 + 8);
        v16h ap = cat8(plo, phi);
        // ---- O += P * V  (Vt contiguous along keys), loads grouped ------
        const _Float16* Vb = Vbase + kb + half * 16;
        v16h bv[4];
#pragma unroll
        for (int j = 0; j < 4; ++j)
            bv[j] = *reinterpret_cast<const v16h*>(Vb + (size_t)(j * 16 + lr) * SEQ);
#pragma unroll
        for (int j = 0; j < 4; ++j)
            acc[j] = wmma_f16(ap, bv[j], acc[j]);
    }

    // ---- normalize and store (token-major f16) --------------------------
#pragma unroll
    for (int r = 0; r < 8; ++r) {
        float inv = 1.0f / l_i[r];
        int qg = q0 + r + 8 * half;
        size_t rowoff = (size_t)(b * SEQ + qg) * DMODEL + h * DHEAD;
#pragma unroll
        for (int j = 0; j < 4; ++j)
            Oh[rowoff + j * 16 + lr] = (_Float16)(acc[j][r] * inv);
    }
}

// ---------------------------------------------------------------------------
extern "C" void kernel_launch(void* const* d_in, const int* in_sizes, int n_in,
                              void* d_out, int out_size, void* d_ws, size_t ws_size,
                              hipStream_t stream) {
    const float* x  = (const float*)d_in[0];
    const int*   tp = (const int*)d_in[1];
    const float* Wq = (const float*)d_in[2];
    const float* Wk = (const float*)d_in[3];
    const float* Wv = (const float*)d_in[4];
    const float* Wo = (const float*)d_in[5];
    float* out = (float*)d_out;

    // Workspace layout (48 MB total, 256B-aligned blocks)
    char* ws = (char*)d_ws;
    size_t off = 0;
    auto alloc = [&](size_t bytes) {
        char* p = ws + off;
        off += (bytes + 255) & ~(size_t)255;
        return p;
    };
    const size_t nTok = (size_t)BS * DMODEL;       // 4M elements
    const size_t nW   = (size_t)DMODEL * DMODEL;   // 1M elements
    _Float16* xh  = (_Float16*)alloc(nTok * 2);
    _Float16* wqh = (_Float16*)alloc(nW * 2);
    _Float16* wkh = (_Float16*)alloc(nW * 2);
    _Float16* wvh = (_Float16*)alloc(nW * 2);
    _Float16* woh = (_Float16*)alloc(nW * 2);
    _Float16* Qh  = (_Float16*)alloc(nTok * 2);
    _Float16* Kh  = (_Float16*)alloc(nTok * 2);
    _Float16* Vt  = (_Float16*)alloc(nTok * 2);
    _Float16* Ah  = (_Float16*)alloc(nTok * 2);

    // 1) fp32 -> fp16 (4 elements/thread)
    cvt_f32_f16_k<<<(int)((nTok / 4 + 255) / 256), 256, 0, stream>>>(x,  xh,  (int)(nTok / 4));
    cvt_f32_f16_k<<<(int)((nW / 4 + 255) / 256),   256, 0, stream>>>(Wq, wqh, (int)(nW / 4));
    cvt_f32_f16_k<<<(int)((nW / 4 + 255) / 256),   256, 0, stream>>>(Wk, wkh, (int)(nW / 4));
    cvt_f32_f16_k<<<(int)((nW / 4 + 255) / 256),   256, 0, stream>>>(Wv, wvh, (int)(nW / 4));
    cvt_f32_f16_k<<<(int)((nW / 4 + 255) / 256),   256, 0, stream>>>(Wo, woh, (int)(nW / 4));

    // 2) QKV projections (WMMA GEMM). grid: M/256 x N/64
    dim3 gg(BS / 256, DMODEL / 64);
    gemm_xwT_k<<<gg, 256, 0, stream>>>(xh, wqh, Qh, nullptr, 0);
    gemm_xwT_k<<<gg, 256, 0, stream>>>(xh, wkh, Kh, nullptr, 0);
    gemm_xwT_k<<<gg, 256, 0, stream>>>(xh, wvh, Vt, nullptr, 1);

    // 3) RoPE on Q and K
    int npairs = BS * (DMODEL / 2);
    rope_k<<<(npairs + 255) / 256, 256, 0, stream>>>(Qh, Kh, tp, npairs);

    // 4) causal flash attention: 4096 wave-tasks, 8 waves/block
    attn_fwd_k<<<512, 256, 0, stream>>>(Qh, Kh, Vt, Ah);

    // 5) output projection (f32 out)
    gemm_xwT_k<<<gg, 256, 0, stream>>>(Ah, woh, nullptr, out, 2);
}
